// VQVAE_86878598463526
// MI455X (gfx1250) — compile-verified, matrix operand order checked by reading, not run on hardware
//
#include <hip/hip_runtime.h>
#include <hip/hip_bf16.h>
#include <stdint.h>

typedef __attribute__((ext_vector_type(16))) _Float16 v16h;
typedef __attribute__((ext_vector_type(8)))  float    v8f;

#define NPIX   65536      // B*H*W = 64*32*32
#define CENC   16
#define LRELU(y) ((y) > 0.f ? (y) : 0.2f * (y))

// ---------------------------------------------------------------------------
// Encoder conv 3x3, cin=1 -> cout=16, NCHW(1ch) in, NHWC out, + bias + lrelu
// ---------------------------------------------------------------------------
__global__ __launch_bounds__(256) void conv3x3_c1(
    const float* __restrict__ in, const float* __restrict__ w,
    const float* __restrict__ bias, float* __restrict__ out)
{
    __shared__ float sw[16 * 9];
    if (threadIdx.x < 144) sw[threadIdx.x] = w[threadIdx.x];
    __syncthreads();

    int n  = blockIdx.x * 256 + threadIdx.x;      // pixel id
    int b  = n >> 10, hw = n & 1023;
    int h  = hw >> 5, x = hw & 31;

    float acc[16];
#pragma unroll
    for (int co = 0; co < 16; ++co) acc[co] = 0.f;

#pragma unroll
    for (int ky = 0; ky < 3; ++ky) {
        int hh = h + ky - 1;
        if (hh < 0 || hh > 31) continue;
#pragma unroll
        for (int kx = 0; kx < 3; ++kx) {
            int xx = x + kx - 1;
            if (xx < 0 || xx > 31) continue;
            float xv = in[(b << 10) + (hh << 5) + xx];
            int tap = ky * 3 + kx;
#pragma unroll
            for (int co = 0; co < 16; ++co)
                acc[co] += xv * sw[co * 9 + tap];
        }
    }
    int base = n << 4;
#pragma unroll
    for (int co = 0; co < 16; ++co) {
        float y = acc[co] + bias[co];
        out[base + co] = LRELU(y);
    }
}

// ---------------------------------------------------------------------------
// Encoder conv 3x3, 16->16, NHWC, + bias + lrelu (+ optional residual add)
// weights OIHW [16,16,3,3] staged in LDS
// ---------------------------------------------------------------------------
__global__ __launch_bounds__(256) void conv3x3_c16(
    const float* __restrict__ in, const float* __restrict__ w,
    const float* __restrict__ bias, const float* __restrict__ res,
    float* __restrict__ out)
{
    __shared__ float sw[16 * 16 * 9];
    for (int i = threadIdx.x; i < 2304; i += 256) sw[i] = w[i];
    __syncthreads();

    int n  = blockIdx.x * 256 + threadIdx.x;
    int b  = n >> 10, hw = n & 1023;
    int h  = hw >> 5, x = hw & 31;

    float acc[16];
#pragma unroll
    for (int co = 0; co < 16; ++co) acc[co] = 0.f;

#pragma unroll
    for (int ky = 0; ky < 3; ++ky) {
        int hh = h + ky - 1;
        if (hh < 0 || hh > 31) continue;
#pragma unroll
        for (int kx = 0; kx < 3; ++kx) {
            int xx = x + kx - 1;
            if (xx < 0 || xx > 31) continue;
            const float* ip = in + (((b << 10) + (hh << 5) + xx) << 4);
            float xi[16];
#pragma unroll
            for (int ci = 0; ci < 16; ++ci) xi[ci] = ip[ci];   // b128-vectorized
            int tap = ky * 3 + kx;
#pragma unroll
            for (int co = 0; co < 16; ++co) {
                const float* wp = &sw[(co << 4) * 9 + tap];
                float s = 0.f;
#pragma unroll
                for (int ci = 0; ci < 16; ++ci) s += xi[ci] * wp[ci * 9];
                acc[co] += s;
            }
        }
    }
    int base = n << 4;
#pragma unroll
    for (int co = 0; co < 16; ++co) {
        float y = acc[co] + bias[co];
        y = LRELU(y);
        if (res) y += res[base + co];
        out[base + co] = y;
    }
}

// ---------------------------------------------------------------------------
// VQ: d2 = |x|^2 + |c|^2 - 2 x.c, first-min argmin (matches jnp.argmin).
// Writes z_e / z_st (== z_e forward value) / z_q in NCHW f32, and z_e as
// f16 rows [N,32] (K padded to 32 with zeros) for the first WMMA GEMM.
// ---------------------------------------------------------------------------
__global__ __launch_bounds__(256) void vq_kernel(
    const float* __restrict__ ze, const float* __restrict__ cb,
    float* __restrict__ ze_o, float* __restrict__ zst_o,
    float* __restrict__ zq_o, _Float16* __restrict__ zeh)
{
    __shared__ float scb[10 * 16];
    __shared__ float sc2[10];
    if (threadIdx.x < 160) scb[threadIdx.x] = cb[threadIdx.x];
    __syncthreads();
    if (threadIdx.x < 10) {
        float s = 0.f;
#pragma unroll
        for (int c = 0; c < 16; ++c) {
            float v = scb[threadIdx.x * 16 + c];
            s += v * v;
        }
        sc2[threadIdx.x] = s;
    }
    __syncthreads();

    int n = blockIdx.x * 256 + threadIdx.x;
    const float* xp = ze + (n << 4);
    float xv[16];
    float sx = 0.f;
#pragma unroll
    for (int c = 0; c < 16; ++c) { xv[c] = xp[c]; sx += xv[c] * xv[c]; }

    float best = 3.402823466e38f;
    int bi = 0;
#pragma unroll
    for (int k = 0; k < 10; ++k) {
        float dot = 0.f;
#pragma unroll
        for (int c = 0; c < 16; ++c) dot += xv[c] * scb[k * 16 + c];
        float d = sx + sc2[k] - 2.f * dot;
        if (d < best) { best = d; bi = k; }
    }

    int b = n >> 10, hw = n & 1023;
    int nb = (b << 14) + hw;                 // NCHW base (c stride = 1024)
#pragma unroll
    for (int c = 0; c < 16; ++c) {
        int o = nb + (c << 10);
        ze_o[o]  = xv[c];
        zst_o[o] = xv[c];                    // straight-through value == z_e
        zq_o[o]  = scb[bi * 16 + c];
    }
    _Float16* zp = zeh + (n << 5);
#pragma unroll
    for (int c = 0; c < 16; ++c) zp[c] = (_Float16)xv[c];
#pragma unroll
    for (int c = 16; c < 32; ++c) zp[c] = (_Float16)0.f;
}

// ---------------------------------------------------------------------------
// Pack torch-layout weight W[O,Kin] (row-major, y = g @ W^T) into per-lane
// WMMA B fragments (32x16 f16 tile). Layout:
//   bp[(((ot*(Kp/32)+kt)*32)+lane)*16 + h]
// lane = {g=lane>>4, n=lane&15};  h = {v=h>>1, lo=h&1}
//   k = kt*32 + (v<4?0:16) + 8*g + 2*(v&3) + lo ;  o = ot*16 + n
// Zero-pads k >= Kin.
// ---------------------------------------------------------------------------
__global__ __launch_bounds__(256) void pack_wmma_b(
    const float* __restrict__ w, _Float16* __restrict__ bp,
    int O, int Kin, int Kp)
{
    int ktiles = Kp >> 5;
    int total  = (O >> 4) * ktiles * 512;
    for (int idx = blockIdx.x * 256 + threadIdx.x; idx < total;
         idx += gridDim.x * 256) {
        int h    = idx & 15;
        int lane = (idx >> 4) & 31;
        int rest = idx >> 9;
        int kt   = rest % ktiles;
        int ot   = rest / ktiles;
        int g    = lane >> 4, nl = lane & 15;
        int v    = h >> 1,    lo = h & 1;
        int k    = (kt << 5) + ((v < 4) ? 0 : 16) + (g << 3) + ((v & 3) << 1) + lo;
        int o    = (ot << 4) + nl;
        float val = (k < Kin) ? w[o * Kin + k] : 0.f;
        bp[idx] = (_Float16)val;
    }
}

// ---------------------------------------------------------------------------
// Decoder GEMM: one wave per 16x16 output tile, K-loop in steps of 32 via
// v_wmma_f32_16x16x32_f16. Epilogue: +bias, then lrelu->f16 (outh) or
// sigmoid->f32 non-temporal (outf, streaming 205MB x_recon past L2).
// ---------------------------------------------------------------------------
__global__ __launch_bounds__(256) void dec_gemm_wmma(
    const _Float16* __restrict__ A, int lda,
    const _Float16* __restrict__ Bp,
    const float* __restrict__ bias,
    _Float16* __restrict__ outh, float* __restrict__ outf,
    int ldo, int Mtiles, int Ntiles, int K)
{
    int wave = threadIdx.x >> 5;
    int lane = threadIdx.x & 31;
    int t = blockIdx.x * 8 + wave;
    if (t >= Mtiles * Ntiles) return;                 // wave-uniform
    int tm = t / Ntiles;
    int tn = t % Ntiles;

    int g  = lane >> 4;
    int ml = lane & 15;
    int row = (tm << 4) + ml;
    const _Float16* arow = A + (size_t)row * lda;

    int ktiles = K >> 5;
    v8f c = {};
    for (int kt = 0; kt < ktiles; ++kt) {
        // A fragment: 16-bit 16x32, lane {g, m}; VGPR v holds halves (k, k+1)
        union { v16h v; uint32_t u[8]; } a;
        int k0 = kt << 5;
#pragma unroll
        for (int v2 = 0; v2 < 8; ++v2) {
            int kb = ((v2 < 4) ? 0 : 16) + (g << 3) + ((v2 & 3) << 1);
            a.u[v2] = *(const uint32_t*)(arow + k0 + kb);
        }
        // B fragment: pre-packed, 16 contiguous halves (32B) per lane
        const v16h b = *(const v16h*)(Bp +
            ((((size_t)tn * ktiles + kt) << 5) + lane) * 16);
        c = __builtin_amdgcn_wmma_f32_16x16x32_f16(
                false, a.v, false, b, (short)0, c, false, false);
    }

    // C/D layout: VGPR r -> m = r + 8*g ; n = lane&15
    int ncol = (tn << 4) + ml;
    float bv = bias[ncol];
#pragma unroll
    for (int r = 0; r < 8; ++r) {
        int m = (tm << 4) + r + (g << 3);
        float y = c[r] + bv;
        if (outh) {
            y = LRELU(y);
            outh[(size_t)m * ldo + ncol] = (_Float16)y;
        } else {
            y = 1.f / (1.f + __expf(-y));
            __builtin_nontemporal_store(y, &outf[(size_t)m * ldo + ncol]);
        }
    }
}

// ---------------------------------------------------------------------------
// Host launcher
// ---------------------------------------------------------------------------
extern "C" void kernel_launch(void* const* d_in, const int* in_sizes, int n_in,
                              void* d_out, int out_size, void* d_ws, size_t ws_size,
                              hipStream_t stream) {
    const float* x    = (const float*)d_in[0];
    const float* cw1  = (const float*)d_in[1];
    const float* cb1  = (const float*)d_in[2];
    const float* cw2  = (const float*)d_in[3];
    const float* cb2  = (const float*)d_in[4];
    const float* r1w1 = (const float*)d_in[5];
    const float* r1b1 = (const float*)d_in[6];
    const float* r1w2 = (const float*)d_in[7];
    const float* r1b2 = (const float*)d_in[8];
    const float* r2w1 = (const float*)d_in[9];
    const float* r2b1 = (const float*)d_in[10];
    const float* r2w2 = (const float*)d_in[11];
    const float* r2b2 = (const float*)d_in[12];
    const float* cbk  = (const float*)d_in[13];
    const float* dw1  = (const float*)d_in[14];
    const float* db1  = (const float*)d_in[15];
    const float* dw2  = (const float*)d_in[16];
    const float* db2  = (const float*)d_in[17];
    const float* dw3  = (const float*)d_in[18];
    const float* db3  = (const float*)d_in[19];

    // output: x_recon [N,784] f32, then z_e, z_st, z_q (NCHW) f32
    float* xrec  = (float*)d_out;
    float* ze_o  = xrec + (size_t)NPIX * 784;
    float* zst_o = ze_o + (size_t)NPIX * CENC;
    float* zq_o  = zst_o + (size_t)NPIX * CENC;

    // workspace carve-up (256B aligned)
    char* ws = (char*)d_ws;
    size_t off = 0;
    auto alloc = [&](size_t bytes) -> char* {
        char* p = ws + off;
        off = (off + bytes + 255) & ~(size_t)255;
        return p;
    };
    float*     hA   = (float*)alloc((size_t)NPIX * 16 * 4);     // 4 MB
    float*     hB   = (float*)alloc((size_t)NPIX * 16 * 4);     // 4 MB
    float*     hC   = (float*)alloc((size_t)NPIX * 16 * 4);     // 4 MB
    _Float16*  zeh  = (_Float16*)alloc((size_t)NPIX * 32 * 2);  // 4 MB (K pad 32)
    _Float16*  act1 = (_Float16*)alloc((size_t)NPIX * 256 * 2); // 32 MB
    _Float16*  act2 = (_Float16*)alloc((size_t)NPIX * 256 * 2); // 32 MB
    _Float16*  w1p  = (_Float16*)alloc((size_t)16 * 1 * 512 * 2);
    _Float16*  w2p  = (_Float16*)alloc((size_t)16 * 8 * 512 * 2);
    _Float16*  w3p  = (_Float16*)alloc((size_t)49 * 8 * 512 * 2);

    dim3 blk(256);

    // weight packing for WMMA B fragments (tiny, overlaps nothing critical)
    pack_wmma_b<<<dim3(32),  blk, 0, stream>>>(dw1, w1p, 256, 16, 32);
    pack_wmma_b<<<dim3(256), blk, 0, stream>>>(dw2, w2p, 256, 256, 256);
    pack_wmma_b<<<dim3(784), blk, 0, stream>>>(dw3, w3p, 784, 256, 256);

    // encoder (NHWC activations, all L2-resident)
    dim3 cg(NPIX / 256);
    conv3x3_c1 <<<cg, blk, 0, stream>>>(x,  cw1,  cb1,  hA);
    conv3x3_c16<<<cg, blk, 0, stream>>>(hA, cw2,  cb2,  nullptr, hB);
    conv3x3_c16<<<cg, blk, 0, stream>>>(hB, r1w1, r1b1, nullptr, hC);
    conv3x3_c16<<<cg, blk, 0, stream>>>(hC, r1w2, r1b2, hB,      hA); // + h2
    conv3x3_c16<<<cg, blk, 0, stream>>>(hA, r2w1, r2b1, nullptr, hC);
    conv3x3_c16<<<cg, blk, 0, stream>>>(hC, r2w2, r2b2, hA,      hB); // + h3

    // VQ + layout conversions (z_e lives in hB NHWC)
    vq_kernel<<<cg, blk, 0, stream>>>(hB, cbk, ze_o, zst_o, zq_o, zeh);

    // decoder: 3 WMMA GEMMs; 8 waves/block, 1 tile/wave
    // L1: [N,32(pad)] x [32,256]   -> act1   (65536 tiles / 8)
    dec_gemm_wmma<<<dim3(8192), blk, 0, stream>>>(
        zeh, 32, w1p, db1, act1, nullptr, 256, NPIX / 16, 16, 32);
    // L2: [N,256] x [256,256]      -> act2
    dec_gemm_wmma<<<dim3(8192), blk, 0, stream>>>(
        act1, 256, w2p, db2, act2, nullptr, 256, NPIX / 16, 16, 256);
    // L3: [N,256] x [256,784]      -> x_recon (sigmoid, NT stores)
    dec_gemm_wmma<<<dim3(25088), blk, 0, stream>>>(
        act2, 256, w3p, db3, nullptr, xrec, 784, NPIX / 16, 49, 256);
}
